// GAT_12833362280659
// MI455X (gfx1250) — matrix operand order repaired and checked by reading, not compiled, not run on hardware
//
#include <hip/hip_runtime.h>
#include <hip/hip_bf16.h>
#include <cstdint>

#define NNODES 20000
#define NEDGES 640000
#define INDIM  131
#define NH     6
#define FD     64
#define HF     384   // NH*FD

typedef __bf16 bf16_t;
typedef bf16_t v16bf __attribute__((ext_vector_type(16)));
typedef float  v8f   __attribute__((ext_vector_type(8)));

// ---------- helpers ----------
__device__ __forceinline__ unsigned short f2bf(float f) {
    unsigned u = __float_as_uint(f);
    unsigned r = u + 0x7FFFu + ((u >> 16) & 1u);   // round-to-nearest-even
    return (unsigned short)(r >> 16);
}
// monotonic float<->uint encoding so atomicMax(u32) == float max, memset(0) == -inf
__device__ __forceinline__ unsigned fenc(float f) {
    unsigned u = __float_as_uint(f);
    return (u & 0x80000000u) ? ~u : (u | 0x80000000u);
}
__device__ __forceinline__ float fdec(unsigned e) {
    unsigned u = (e & 0x80000000u) ? (e & 0x7FFFFFFFu) : ~e;
    return __uint_as_float(u);
}

// ---------- conversion kernels ----------
__global__ void cvt_feat(const float* __restrict__ x, unsigned short* __restrict__ xb,
                         int n, int din, int kpad) {
    int idx = blockIdx.x * blockDim.x + threadIdx.x;
    if (idx >= n * kpad) return;
    int r = idx / kpad, k = idx - r * kpad;
    xb[idx] = (k < din) ? f2bf(x[(size_t)r * din + k]) : (unsigned short)0;
}

// W [din x HF] row-major  ->  wbT [HF x kpad] (column-major B, zero-padded K)
__global__ void cvt_wT(const float* __restrict__ w, unsigned short* __restrict__ wbT,
                       int din, int kpad) {
    int idx = blockIdx.x * blockDim.x + threadIdx.x;
    if (idx >= HF * kpad) return;
    int c = idx / kpad, k = idx - c * kpad;
    wbT[idx] = (k < din) ? f2bf(w[(size_t)k * HF + c]) : (unsigned short)0;
}

// ---------- WMMA GEMM with async global->LDS staging of B ----------
// C[M x HF] = A[M x kpad](bf16) * B(kpad x HF), B given as BT[HF x kpad].
// Block = 8 waves = 128(M) x 64(N) tile. B window (64 cols x 64 K, 8 KB) staged in
// LDS via global_load_async_to_lds_b128 (ASYNCcnt), consumed with ds_load_b128.
__global__ __launch_bounds__(256) void gemm_wmma(const unsigned short* __restrict__ A,
                                                 const unsigned short* __restrict__ BT,
                                                 float* __restrict__ C,
                                                 int M, int kpad) {
    __shared__ unsigned short sb[64 * 64];          // [col][k] bf16, 8 KB
    const int mtiles = M >> 4;                      // 1250
    const int ntiles = HF >> 6;                     // 6
    const int nt = blockIdx.x % ntiles;
    const int mb = blockIdx.x / ntiles;
    int mt = mb * 8 + (threadIdx.x >> 5);
    if (mt >= mtiles) mt = mtiles - 1;              // clamp (duplicate work; barrier-safe)
    const int lane = threadIdx.x & 31;
    const int l15  = lane & 15;
    const int hi   = lane >> 4;
    const int kbA  = hi << 4;                       // A: K split at granularity 16
    const int kbB  = hi << 4;                       // B: K split at granularity 16

    const unsigned short* arow = A + (size_t)((mt << 4) + l15) * kpad;
    const int col0 = nt << 6;

    // staging: thread t copies 32 B (16 bf16) : col = t&63, K-segment = (t>>6)*16
    const int scol = threadIdx.x & 63;
    const int sseg = (threadIdx.x >> 6) << 4;
    const unsigned lds_off = (unsigned)(uintptr_t)&sb[scol * 64 + sseg];
    const unsigned short* gsrc = BT + (size_t)(col0 + scol) * kpad + sseg;

    v8f acc0 = {}, acc1 = {}, acc2 = {}, acc3 = {};
    for (int kk = 0; kk < kpad; kk += 64) {
        __syncthreads();                             // LDS reuse guard
        uint64_t ga = (uint64_t)(const void*)(gsrc + kk);
        asm volatile("global_load_async_to_lds_b128 %0, %1, off"
                     :: "v"(lds_off), "v"(ga) : "memory");
        asm volatile("global_load_async_to_lds_b128 %0, %1, off"
                     :: "v"(lds_off + 16u), "v"(ga + 16) : "memory");
        asm volatile("s_wait_asynccnt 0x0" ::: "memory");
        __syncthreads();                             // data visible block-wide
        #pragma unroll
        for (int c = 0; c < 2; ++c) {
            const int k0 = kk + c * 32;
            union { uint4 u[2]; v16bf v; } af, b0, b1, b2, b3;
            af.u[0] = *(const uint4*)(arow + k0 + kbA);       // K = kbA..kbA+7
            af.u[1] = *(const uint4*)(arow + k0 + kbA + 8);   // K = kbA+8..+15
            const unsigned short* sbase = sb + c * 32 + kbB;  // [col][k] in LDS
            b0.u[0] = *(const uint4*)(sbase + (l15 +  0) * 64);
            b0.u[1] = *(const uint4*)(sbase + (l15 +  0) * 64 + 8);
            b1.u[0] = *(const uint4*)(sbase + (l15 + 16) * 64);
            b1.u[1] = *(const uint4*)(sbase + (l15 + 16) * 64 + 8);
            b2.u[0] = *(const uint4*)(sbase + (l15 + 32) * 64);
            b2.u[1] = *(const uint4*)(sbase + (l15 + 32) * 64 + 8);
            b3.u[0] = *(const uint4*)(sbase + (l15 + 48) * 64);
            b3.u[1] = *(const uint4*)(sbase + (l15 + 48) * 64 + 8);
            acc0 = __builtin_amdgcn_wmma_f32_16x16x32_bf16(false, af.v, false, b0.v, (short)0, acc0, false, false);
            acc1 = __builtin_amdgcn_wmma_f32_16x16x32_bf16(false, af.v, false, b1.v, (short)0, acc1, false, false);
            acc2 = __builtin_amdgcn_wmma_f32_16x16x32_bf16(false, af.v, false, b2.v, (short)0, acc2, false, false);
            acc3 = __builtin_amdgcn_wmma_f32_16x16x32_bf16(false, af.v, false, b3.v, (short)0, acc3, false, false);
        }
    }
    // C/D layout: lane&15 = N col, VGPR i -> M = i + 8*(lane>>4)
    float* crow = C + (size_t)((mt << 4) + (hi << 3)) * HF + (nt << 6) + l15;
    #pragma unroll
    for (int i = 0; i < 8; ++i) {
        crow[(size_t)i * HF +  0] = acc0[i];
        crow[(size_t)i * HF + 16] = acc1[i];
        crow[(size_t)i * HF + 32] = acc2[i];
        crow[(size_t)i * HF + 48] = acc3[i];
    }
}

// ---------- attention logits: el/er [N,H] ----------
__global__ void logits_k(const float* __restrict__ ft, const float* __restrict__ al,
                         const float* __restrict__ ar, float* __restrict__ el,
                         float* __restrict__ er) {
    int idx = blockIdx.x * blockDim.x + threadIdx.x;
    if (idx >= NNODES * NH) return;
    int n = idx / NH, h = idx - n * NH;
    const float4* f4 = (const float4*)(ft + (size_t)n * HF + h * FD);
    const float4* l4 = (const float4*)(al + h * FD);
    const float4* r4 = (const float4*)(ar + h * FD);
    float sl = 0.f, sr = 0.f;
    #pragma unroll
    for (int i = 0; i < FD / 4; ++i) {
        float4 f = f4[i], l = l4[i], r = r4[i];
        sl += f.x * l.x + f.y * l.y + f.z * l.z + f.w * l.w;
        sr += f.x * r.x + f.y * r.y + f.z * r.z + f.w * r.w;
    }
    el[idx] = sl; er[idx] = sr;
}

// ---------- edge softmax: pass 1 (segment max via monotonic u32 atomicMax) ----------
__global__ void edge_max_k(const int* __restrict__ src, const int* __restrict__ dst,
                           const float* __restrict__ el, const float* __restrict__ er,
                           unsigned* __restrict__ emax) {
    int e = blockIdx.x * blockDim.x + threadIdx.x;
    if (e >= NEDGES) return;
    int s = src[e], d = dst[e];
    #pragma unroll
    for (int h = 0; h < NH; ++h) {
        float v = el[s * NH + h] + er[d * NH + h];
        v = v > 0.f ? v : 0.2f * v;                       // leaky relu
        atomicMax(&emax[d * NH + h], fenc(v));
    }
}

// ---------- edge softmax: pass 2 (exp + segment sum) ----------
__global__ void edge_exp_k(const int* __restrict__ src, const int* __restrict__ dst,
                           const float* __restrict__ el, const float* __restrict__ er,
                           const unsigned* __restrict__ emax, float* __restrict__ ee,
                           float* __restrict__ denom) {
    int e = blockIdx.x * blockDim.x + threadIdx.x;
    if (e >= NEDGES) return;
    int s = src[e], d = dst[e];
    #pragma unroll
    for (int h = 0; h < NH; ++h) {
        float v = el[s * NH + h] + er[d * NH + h];
        v = v > 0.f ? v : 0.2f * v;
        unsigned enc = emax[d * NH + h];
        float m = (enc == 0u) ? 0.f : fdec(enc);          // untouched -> zero in-degree -> 0
        float x = __expf(v - m);
        ee[(size_t)e * NH + h] = x;
        atomicAdd(&denom[d * NH + h], x);
    }
}

// ---------- aggregation: one wave per edge, atomicAdd into agg[dst] ----------
__global__ __launch_bounds__(256) void edge_agg_k(const int* __restrict__ src,
                                                  const int* __restrict__ dst,
                                                  const float* __restrict__ ft,
                                                  const float* __restrict__ ee,
                                                  const float* __restrict__ denom,
                                                  float* __restrict__ agg) {
    int wid = blockIdx.x * (blockDim.x >> 5) + (threadIdx.x >> 5);
    if (wid >= NEDGES) return;
    int lane = threadIdx.x & 31;
    int s = src[wid], d = dst[wid];
    #pragma unroll
    for (int h = 0; h < NH; ++h) {
        float a = ee[(size_t)wid * NH + h] / fmaxf(denom[d * NH + h], 1e-9f);
        float2 v = ((const float2*)(ft + (size_t)s * HF + h * FD))[lane];
        float* o = agg + (size_t)d * HF + h * FD + lane * 2;
        atomicAdd(o,     v.x * a);
        atomicAdd(o + 1, v.y * a);
    }
}

// ---------- bias + optional ELU ----------
__global__ void bias_act_k(const float* __restrict__ agg, const float* __restrict__ b,
                           float* __restrict__ out, int apply_act) {
    int idx = blockIdx.x * blockDim.x + threadIdx.x;
    if (idx >= NNODES * HF) return;
    int c = idx % HF;
    float v = agg[idx] + b[c];
    if (apply_act) v = v > 0.f ? v : (__expf(v) - 1.f);   // ELU
    out[idx] = v;
}

// ---------- final: mean over heads ----------
__global__ void mean_heads_k(const float* __restrict__ agg, const float* __restrict__ b,
                             float* __restrict__ out) {
    int idx = blockIdx.x * blockDim.x + threadIdx.x;
    if (idx >= NNODES * FD) return;
    int n = idx / FD, f = idx - n * FD;
    float s = 0.f;
    #pragma unroll
    for (int h = 0; h < NH; ++h) s += agg[(size_t)n * HF + h * FD + f] + b[h * FD + f];
    out[idx] = s * (1.f / 6.f);
}

// ---------------------------------------------------------------------------
extern "C" void kernel_launch(void* const* d_in, const int* in_sizes, int n_in,
                              void* d_out, int out_size, void* d_ws, size_t ws_size,
                              hipStream_t stream) {
    (void)in_sizes; (void)n_in; (void)out_size; (void)ws_size;
    const float* nfeat = (const float*)d_in[0];
    const int*   src   = (const int*)d_in[1];
    const int*   dst   = (const int*)d_in[2];

    uint8_t* ws = (uint8_t*)d_ws;
    size_t off = 0;
    auto take = [&](size_t bytes) -> void* {
        void* p = ws + off;
        off += (bytes + 255) & ~(size_t)255;
        return p;
    };
    float*          fX   = (float*)take((size_t)NNODES * HF * 4);   // layer activations
    float*          fFT  = (float*)take((size_t)NNODES * HF * 4);   // projected features
    float*          fAG  = (float*)take((size_t)NNODES * HF * 4);   // aggregation accum
    unsigned short* hbf  = (unsigned short*)take((size_t)NNODES * HF * 2);
    unsigned short* wbf  = (unsigned short*)take((size_t)HF * HF * 2);
    float*          el   = (float*)take((size_t)NNODES * NH * 4);
    float*          er   = (float*)take((size_t)NNODES * NH * 4);
    unsigned*       emax = (unsigned*)take((size_t)NNODES * NH * 4);
    float*          den  = (float*)take((size_t)NNODES * NH * 4);
    float*          ee   = (float*)take((size_t)NEDGES * NH * 4);

    const int mtiles = NNODES / 16;                       // 1250
    const int ntiles = HF / 64;                           // 6
    const int gemm_blocks = ((mtiles + 7) / 8) * ntiles;  // 157*6
    const int edge_blocks = (NEDGES + 255) / 256;
    const int agg_blocks  = (NEDGES + 7) / 8;

    for (int l = 0; l < 4; ++l) {
        const int din  = (l == 0) ? INDIM : HF;
        const int kpad = (din + 63) & ~63;                // multiple of 64 (2 WMMA chunks)
        const float* x  = (l == 0) ? nfeat : fX;
        const float* W  = (const float*)d_in[3 + 4 * l];
        const float* al = (const float*)d_in[4 + 4 * l];
        const float* ar = (const float*)d_in[5 + 4 * l];
        const float* bb = (const float*)d_in[6 + 4 * l];

        cvt_feat<<<(NNODES * kpad + 255) / 256, 256, 0, stream>>>(x, hbf, NNODES, din, kpad);
        cvt_wT<<<(HF * kpad + 255) / 256, 256, 0, stream>>>(W, wbf, din, kpad);
        gemm_wmma<<<gemm_blocks, 256, 0, stream>>>(hbf, wbf, fFT, NNODES, kpad);

        logits_k<<<(NNODES * NH + 255) / 256, 256, 0, stream>>>(fFT, al, ar, el, er);

        hipMemsetAsync(emax, 0, (size_t)NNODES * NH * 4, stream);
        hipMemsetAsync(den,  0, (size_t)NNODES * NH * 4, stream);
        hipMemsetAsync(fAG,  0, (size_t)NNODES * HF * 4, stream);

        edge_max_k<<<edge_blocks, 256, 0, stream>>>(src, dst, el, er, emax);
        edge_exp_k<<<edge_blocks, 256, 0, stream>>>(src, dst, el, er, emax, ee, den);
        edge_agg_k<<<agg_blocks, 256, 0, stream>>>(src, dst, fFT, ee, den, fAG);

        if (l < 3) {
            bias_act_k<<<(NNODES * HF + 255) / 256, 256, 0, stream>>>(fAG, bb, fX, 1);
        } else {
            mean_heads_k<<<(NNODES * FD + 255) / 256, 256, 0, stream>>>(fAG, bb, (float*)d_out);
        }
    }
}
